// InvariantPolynomial_90993177133393
// MI455X (gfx1250) — compile-verified, matrix operand order checked by reading, not run on hardware
//
#include <hip/hip_runtime.h>
#include <cmath>
#include <complex>
#include <vector>
#include <algorithm>

// ---------------------------------------------------------------------------
// Types for CDNA5 WMMA
// ---------------------------------------------------------------------------
typedef __attribute__((ext_vector_type(16))) _Float16 v16h;
typedef __attribute__((ext_vector_type(8)))  float    v8f;

#define RS6     0.40824829f        // 1/sqrt(6)
#define ALPHA2  0.37796447f        // sqrt(5/35)
#define INV_S20 0.22360680f        // 1/sqrt(20)
#define INV_S30 0.18257419f        // 1/sqrt(30)
#define SILU_N  1.679177f

// Native-rate helpers: v_exp_f32 / v_rcp_f32 on the TRANS pipe (co-executes
// with VALU; WMMA is also TRANS-tracked, so these overlap with matrix ops).
__device__ __forceinline__ float fexp(float x)  { return __expf(x); }
__device__ __forceinline__ float frcp(float x)  { return __builtin_amdgcn_rcpf(x); }

// ---------------------------------------------------------------------------
// Spherical harmonics (l = 0..3), exactly matching the reference ordering.
// Returns |v| and fills sh[16] using the normalized direction.
// ---------------------------------------------------------------------------
__device__ __forceinline__ float compute_sh(float vx, float vy, float vz, float* sh) {
    float d2 = vx * vx + vy * vy + vz * vz;
    float d  = sqrtf(d2);
    float inv = d > 0.f ? frcp(d) : 0.f;
    float X = vx * inv, Y = vy * inv, Z = vz * inv;
    float x2 = X * X, y2 = Y * Y, z2 = Z * Z;
    const float s3 = 1.7320508f, s5 = 2.2360680f, s15 = 3.8729833f;
    const float s70_4 = 2.0916501f, s105 = 10.246951f, s42_4 = 1.6201852f;
    const float s7_2 = 1.3228757f, s105_2 = 5.1234754f;
    sh[0]  = 1.f;
    sh[1]  = s3 * X;  sh[2] = s3 * Y;  sh[3] = s3 * Z;
    sh[4]  = s15 * X * Z;
    sh[5]  = s15 * X * Y;
    sh[6]  = s5 * (y2 - 0.5f * (x2 + z2));
    sh[7]  = s15 * Y * Z;
    sh[8]  = 0.5f * s15 * (z2 - x2);
    sh[9]  = s70_4 * X * (3.f * z2 - x2);
    sh[10] = s105 * X * Y * Z;
    sh[11] = s42_4 * X * (5.f * y2 - 1.f);
    sh[12] = s7_2 * Y * (5.f * y2 - 3.f);
    sh[13] = s42_4 * Z * (5.f * y2 - 1.f);
    sh[14] = s105_2 * Y * (z2 - x2);
    sh[15] = s70_4 * Z * (z2 - 3.f * x2);
    return d;
}

// Radial window u(t) = t>0 ? exp(-1/t) : 0, via native rcp+exp.
__device__ __forceinline__ float uwin(float t) {
    return t > 0.f ? fexp(-frcp(t)) : 0.f;
}

// ---------------------------------------------------------------------------
// Kernel 1: per-edge geometry + radial embedding, then the two-layer MLP
// done with v_wmma_f32_16x16x32_f16 (16 edges per M-tile, K padded to 32),
// then the first tensor product scattered into node_mid with f32 atomics
// (node_mid is 16 MB -> resident in the 192 MB L2), plus bincount(edge_to).
// Block = 256 threads = 8 wave32; each wave owns 2 M-tiles (32 edges).
// ---------------------------------------------------------------------------
__global__ void __launch_bounds__(256)
edge_forward_kernel(const float* __restrict__ x, const float* __restrict__ pos,
                    const int* __restrict__ ef, const int* __restrict__ et,
                    const float* __restrict__ W1, const float* __restrict__ W2,
                    float* __restrict__ node_mid, int* __restrict__ counts, int E) {
    __shared__ _Float16 sA[256][32];   // emb tile (f16, K padded to 32)
    __shared__ _Float16 sH[256][32];   // hidden tile (f16, 30 padded to 32)
    __shared__ _Float16 sW1[32][32];   // W1 transposed: [n][k]
    __shared__ _Float16 sW2[32][32];   // W2 transposed: [n][k]
    __shared__ float    sT[256][20];   // tp_w output (f32)

    const int tid  = threadIdx.x;
    const int lane = tid & 31;
    const int wid  = tid >> 5;
    const int e    = blockIdx.x * 256 + tid;
    const bool valid = (e < E);
    const int ec = valid ? e : 0;

    // Stage transposed f16 weights into LDS (zero-padded to 32x32).
    for (int idx = tid; idx < 1024; idx += 256) {
        int n = idx >> 5, k = idx & 31;
        sW1[n][k] = (_Float16)((k < 20 && n < 30) ? W1[k * 30 + n] : 0.f);
        sW2[n][k] = (_Float16)((k < 30 && n < 20) ? W2[k * 20 + n] : 0.f);
    }

    // Prefetch upcoming edge indices (emits global_prefetch_b8).
    __builtin_prefetch((const void*)(ef + e + 4096), 0, 1);
    __builtin_prefetch((const void*)(et + e + 4096), 0, 1);

    const int src = ef[ec], dst = et[ec];
    float sh[16];
    const float d = compute_sh(pos[dst * 3 + 0] - pos[src * 3 + 0],
                               pos[dst * 3 + 1] - pos[src * 3 + 1],
                               pos[dst * 3 + 2] - pos[src * 3 + 2], sh);
    const float s_feat = x[dst];

    // Radial soft-one-hot embedding (BASIS=20, R_MAX=3.5).
    {
        const float step = 3.5f / 21.f;
        const float istep = 6.f;             // 1/step
        const float coef = 1.14136f * fexp(2.f);
        #pragma unroll
        for (int i = 0; i < 32; i++) {
            float v = 0.f;
            if (i < 20) {
                float diff = (d - (float)(i + 1) * step) * istep;
                v = coef * uwin(diff + 1.f) * uwin(1.f - diff);
            }
            sA[tid][i] = (_Float16)((valid && i < 20) ? v : 0.f);
        }
    }
    __syncthreads();

    // ----- GEMM1: h = NORM * silu(emb @ W1 / sqrt(20)) via WMMA -------------
    const int lrow = lane & 15;
    const int khi  = (lane < 16) ? 0 : 8;   // K sub-block per lane half
    #pragma unroll
    for (int mt = 0; mt < 2; mt++) {
        const int arow = (wid * 2 + mt) * 16 + lrow;
        v16h a;
        #pragma unroll
        for (int v = 0; v < 8; v++) {
            int k0 = ((v < 4) ? 2 * v : 2 * v + 8) + khi;  // {0..7,16..23}(+8)
            a[2 * v]     = sA[arow][k0];
            a[2 * v + 1] = sA[arow][k0 + 1];
        }
        #pragma unroll
        for (int nt = 0; nt < 2; nt++) {
            const int col = nt * 16 + lrow;
            v16h b;
            #pragma unroll
            for (int v = 0; v < 8; v++) {
                int k0 = ((v < 4) ? 2 * v : 2 * v + 8) + khi;
                b[2 * v]     = sW1[col][k0];
                b[2 * v + 1] = sW1[col][k0 + 1];
            }
            v8f c = {};
            c = __builtin_amdgcn_wmma_f32_16x16x32_f16(false, a, false, b,
                                                       (short)0, c, false, false);
            #pragma unroll
            for (int r = 0; r < 8; r++) {
                const int orow = (wid * 2 + mt) * 16 + r + khi;
                float val = c[r] * INV_S20;
                // silu(v)*norm = norm * v / (1 + e^-v), native exp + rcp
                val = SILU_N * val * frcp(1.f + fexp(-val));
                sH[orow][col] = (_Float16)val;             // padded cols -> 0
            }
        }
    }
    __syncthreads();

    // ----- GEMM2: tp_w = h @ W2 / sqrt(30) via WMMA -------------------------
    #pragma unroll
    for (int mt = 0; mt < 2; mt++) {
        const int arow = (wid * 2 + mt) * 16 + lrow;
        v16h a;
        #pragma unroll
        for (int v = 0; v < 8; v++) {
            int k0 = ((v < 4) ? 2 * v : 2 * v + 8) + khi;
            a[2 * v]     = sH[arow][k0];
            a[2 * v + 1] = sH[arow][k0 + 1];
        }
        #pragma unroll
        for (int nt = 0; nt < 2; nt++) {
            const int col = nt * 16 + lrow;
            v16h b;
            #pragma unroll
            for (int v = 0; v < 8; v++) {
                int k0 = ((v < 4) ? 2 * v : 2 * v + 8) + khi;
                b[2 * v]     = sW2[col][k0];
                b[2 * v + 1] = sW2[col][k0 + 1];
            }
            v8f c = {};
            c = __builtin_amdgcn_wmma_f32_16x16x32_f16(false, a, false, b,
                                                       (short)0, c, false, false);
            #pragma unroll
            for (int r = 0; r < 8; r++) {
                const int orow = (wid * 2 + mt) * 16 + r + khi;
                if (col < 20) sT[orow][col] = c[r] * INV_S30;
            }
        }
    }
    __syncthreads();

    // ----- TP1: mid = s * tp_w[l,u] * sh_block[l], scatter to node_mid ------
    if (valid) {
        float* base = node_mid + (size_t)src * 80;
        int off = 0;
        #pragma unroll
        for (int l = 0; l < 4; l++) {
            const int wl = 2 * l + 1;
            const int so = l * l;
            #pragma unroll
            for (int u = 0; u < 5; u++) {
                const float cc = s_feat * sT[tid][l * 5 + u] * RS6;
                for (int k = 0; k < wl; k++)
                    atomicAdd(base + off + u * wl + k, cc * sh[so + k]);
            }
            off += 5 * wl;
        }
        atomicAdd(&counts[dst], 1);   // bincount(edge_to)
    }
}

// ---------------------------------------------------------------------------
// Kernel 2: second tensor product. Gather node_mid[edge_to] (80 f32, L2-
// resident), contract with Wigner-3j over the 7 paths, scatter 5 outputs
// into node_out with f32 atomics.
// ---------------------------------------------------------------------------
__global__ void __launch_bounds__(256)
edge_tp2_kernel(const float* __restrict__ pos,
                const int* __restrict__ ef, const int* __restrict__ et,
                const float* __restrict__ tp2w, const float* __restrict__ w3j,
                const float* __restrict__ node_mid, float* __restrict__ node_out,
                int E) {
    const int e = blockIdx.x * blockDim.x + threadIdx.x;
    if (e >= E) return;
    __builtin_prefetch((const void*)(et + e + 4096), 0, 1);

    const int src = ef[e], dst = et[e];
    float sh[16];
    (void)compute_sh(pos[dst * 3 + 0] - pos[src * 3 + 0],
                     pos[dst * 3 + 1] - pos[src * 3 + 1],
                     pos[dst * 3 + 2] - pos[src * 3 + 2], sh);

    float mm[80];
    const float* m = node_mid + (size_t)dst * 80;
    #pragma unroll
    for (int i = 0; i < 80; i++) mm[i] = m[i];

    const int L1[7]   = {0, 1, 1, 2, 2, 3, 3};
    const int L2[7]   = {2, 1, 3, 0, 2, 1, 3};
    const int WOFF[7] = {0, 25, 70, 175, 200, 325, 430};
    const int MOFF[4] = {0, 5, 20, 45};

    float acc[5] = {0.f, 0.f, 0.f, 0.f, 0.f};
    #pragma unroll
    for (int p = 0; p < 7; p++) {
        const int l1 = L1[p], l2 = L2[p];
        const int w1 = 2 * l1 + 1, w2 = 2 * l2 + 1;
        float mu[7];
        for (int i = 0; i < w1; i++) {
            float t = 0.f;
            #pragma unroll
            for (int u = 0; u < 5; u++) t += mm[MOFF[l1] + u * w1 + i] * tp2w[p * 5 + u];
            mu[i] = t;
        }
        const float* W = w3j + WOFF[p];
        for (int i = 0; i < w1; i++) {
            for (int j = 0; j < w2; j++) {
                const float f = mu[i] * sh[l2 * l2 + j];
                const float* Wr = W + (i * w2 + j) * 5;
                #pragma unroll
                for (int k = 0; k < 5; k++) acc[k] += Wr[k] * f;
            }
        }
    }
    const float scl = ALPHA2 * RS6;
    #pragma unroll
    for (int k = 0; k < 5; k++)
        atomicAdd(&node_out[(size_t)src * 5 + k], scl * acc[k]);
}

// ---------------------------------------------------------------------------
// Kernel 3: argmax(bincount) via packed 64-bit atomicMax.
// High 32 bits = count, low 32 bits = ~idx -> first-max (smallest idx) wins.
// ---------------------------------------------------------------------------
__global__ void argmax_kernel(const int* __restrict__ counts,
                              unsigned long long* __restrict__ best, int N) {
    const int i = blockIdx.x * blockDim.x + threadIdx.x;
    if (i < N) {
        unsigned long long v = ((unsigned long long)(unsigned)counts[i] << 32)
                             | (unsigned long long)(0xFFFFFFFFu - (unsigned)i);
        atomicMax(best, v);
    }
}

__global__ void finalize_kernel(const unsigned long long* __restrict__ best,
                                const float* __restrict__ node_out,
                                float* __restrict__ out) {
    unsigned long long b = *best;
    unsigned idx = 0xFFFFFFFFu - (unsigned)(b & 0xFFFFFFFFull);
    if ((b >> 32) == 0ull) idx = 0;
    const int k = threadIdx.x;
    if (k < 5) out[k] = node_out[(size_t)idx * 5 + k];
}

// ---------------------------------------------------------------------------
// Host-side Wigner-3j generation (exact port of the reference math).
// ---------------------------------------------------------------------------
static double h_fact(int n) { double r = 1.0; for (int i = 2; i <= n; i++) r *= i; return r; }

static double h_su2_cg(int j1, int m1, int j2, int m2, int j3, int m3) {
    if (m3 != m1 + m2) return 0.0;
    int vmin = std::max({-j1 + j2 + m3, -j1 + m1, 0});
    int vmax = std::min({j2 + j3 + m1, j3 - j1 + j2, j3 + m3});
    double C = std::sqrt((2.0 * j3 + 1) * h_fact(j3 + j1 - j2) * h_fact(j3 - j1 + j2) *
                         h_fact(j1 + j2 - j3) * h_fact(j3 + m3) * h_fact(j3 - m3) /
                         (h_fact(j1 + j2 + j3 + 1) * h_fact(j1 - m1) * h_fact(j1 + m1) *
                          h_fact(j2 - m2) * h_fact(j2 + m2)));
    double S = 0.0;
    for (int v = vmin; v <= vmax; v++) {
        double sgn = ((v + j2 + m2) % 2) ? -1.0 : 1.0;
        S += sgn / h_fact(v) * h_fact(j2 + j3 + m1 - v) * h_fact(j1 - m1 + v) /
             h_fact(j3 - j1 + j2 - v) / h_fact(j3 + m3 - v) / h_fact(v + j1 - j2 - m3);
    }
    return C * S;
}

static void h_real_basis(int l, std::complex<double>* q) {
    const int n = 2 * l + 1;
    std::fill(q, q + n * n, std::complex<double>(0, 0));
    const double r = M_SQRT1_2;
    for (int m = -l; m < 0; m++) {
        q[(l + m) * n + (l - m)] = std::complex<double>(r, 0);   // col l+|m|
        q[(l + m) * n + (l + m)] = std::complex<double>(0, -r);  // col l-|m|
    }
    q[l * n + l] = std::complex<double>(1, 0);
    for (int m = 1; m <= l; m++) {
        double sgn = (m % 2) ? -1.0 : 1.0;
        q[(l + m) * n + (l + m)] = std::complex<double>(sgn * r, 0);
        q[(l + m) * n + (l - m)] = std::complex<double>(0, sgn * r);
    }
    std::complex<double> f = std::pow(std::complex<double>(0, -1), l);
    for (int i = 0; i < n * n; i++) q[i] *= f;
}

static void compute_w3j_host(float* out) {
    const int L1[7] = {0, 1, 1, 2, 2, 3, 3};
    const int L2[7] = {2, 1, 3, 0, 2, 1, 3};
    int off = 0;
    for (int p = 0; p < 7; p++) {
        const int l1 = L1[p], l2 = L2[p], l3 = 2;
        const int n1 = 2 * l1 + 1, n2 = 2 * l2 + 1, n3 = 2 * l3 + 1;
        std::vector<double> C((size_t)n1 * n2 * n3, 0.0);
        for (int m1 = -l1; m1 <= l1; m1++)
            for (int m2 = -l2; m2 <= l2; m2++)
                if (std::abs(m1 + m2) <= l3)
                    C[((size_t)(l1 + m1) * n2 + (l2 + m2)) * n3 + (l3 + m1 + m2)] =
                        h_su2_cg(l1, m1, l2, m2, l3, m1 + m2);
        std::vector<std::complex<double>> Q1((size_t)n1 * n1), Q2((size_t)n2 * n2),
                                          Q3((size_t)n3 * n3);
        h_real_basis(l1, Q1.data());
        h_real_basis(l2, Q2.data());
        h_real_basis(l3, Q3.data());
        // Cr[j,l,m] = sum_{i,k,n} Q1[i,j] Q2[k,l] conj(Q3[n,m]) C[i,k,n]
        std::vector<std::complex<double>> Cr((size_t)n1 * n2 * n3, {0, 0});
        for (int i = 0; i < n1; i++)
            for (int j = 0; j < n1; j++)
                for (int k = 0; k < n2; k++)
                    for (int l = 0; l < n2; l++)
                        for (int nn = 0; nn < n3; nn++)
                            for (int m = 0; m < n3; m++)
                                Cr[((size_t)j * n2 + l) * n3 + m] +=
                                    Q1[(size_t)i * n1 + j] * Q2[(size_t)k * n2 + l] *
                                    std::conj(Q3[(size_t)nn * n3 + m]) *
                                    C[((size_t)i * n2 + k) * n3 + nn];
        double nrm = 0.0;
        for (auto& z : Cr) nrm += z.real() * z.real();
        nrm = std::sqrt(nrm);
        const double inv = nrm > 0.0 ? 1.0 / nrm : 1.0;
        for (int t = 0; t < n1 * n2 * n3; t++) out[off + t] = (float)(Cr[t].real() * inv);
        off += n1 * n2 * n3;
    }
}

// ---------------------------------------------------------------------------
// Launch
// ---------------------------------------------------------------------------
extern "C" void kernel_launch(void* const* d_in, const int* in_sizes, int n_in,
                              void* d_out, int out_size, void* d_ws, size_t ws_size,
                              hipStream_t stream) {
    const float* x    = (const float*)d_in[0];
    const float* pos  = (const float*)d_in[1];
    const int*   ef   = (const int*)d_in[2];
    const int*   et   = (const int*)d_in[3];
    const float* W1   = (const float*)d_in[4];
    const float* W2   = (const float*)d_in[5];
    const float* tp2w = (const float*)d_in[6];
    float* out = (float*)d_out;

    const int N = in_sizes[0];
    const int E = in_sizes[2];

    // Workspace layout
    size_t off_node_mid = 0;
    size_t off_node_out = off_node_mid + (size_t)N * 80 * sizeof(float);
    size_t off_counts   = off_node_out + (size_t)N * 5 * sizeof(float);
    size_t off_best     = (off_counts + (size_t)N * sizeof(int) + 7) & ~(size_t)7;
    size_t off_w3j      = off_best + 8;

    float* node_mid = (float*)((char*)d_ws + off_node_mid);
    float* node_out = (float*)((char*)d_ws + off_node_out);
    int*   counts   = (int*)((char*)d_ws + off_counts);
    unsigned long long* best = (unsigned long long*)((char*)d_ws + off_best);
    float* w3j      = (float*)((char*)d_ws + off_w3j);

    // Recomputed every call (deterministic; tiny), uploaded async.
    static float w3j_host[675];
    compute_w3j_host(w3j_host);

    hipMemsetAsync(d_ws, 0, off_w3j, stream);
    hipMemcpyAsync(w3j, w3j_host, 675 * sizeof(float), hipMemcpyHostToDevice, stream);

    const int eb = (E + 255) / 256;
    const int nb = (N + 255) / 256;
    edge_forward_kernel<<<eb, 256, 0, stream>>>(x, pos, ef, et, W1, W2,
                                                node_mid, counts, E);
    edge_tp2_kernel<<<eb, 256, 0, stream>>>(pos, ef, et, tp2w, w3j,
                                            node_mid, node_out, E);
    argmax_kernel<<<nb, 256, 0, stream>>>(counts, best, N);
    finalize_kernel<<<1, 32, 0, stream>>>(best, node_out, out);
}
